// MoeMLP_35759897706716
// MI455X (gfx1250) — matrix operand order repaired and checked by reading, not compiled, not run on hardware
//
#include <hip/hip_runtime.h>

// ---------------- problem constants ----------------
#define S_DIM 2048
#define B_DIM 2
#define H_DIM 1024
#define F_DIM 2048
#define E_DIM 8
#define K_TOP 2
#define T_DIM (S_DIM * B_DIM)      // 4096 tokens
#define NPAIR (T_DIM * K_TOP)      // 8192 (token,expert) slots

// ---------------- CDNA5 async global->LDS path (guarded) ----------------
#if defined(__has_builtin)
#  if __has_builtin(__builtin_amdgcn_global_load_async_to_lds_b128)
#    define USE_ASYNC_LDS 1
#  endif
#  if __has_builtin(__builtin_amdgcn_s_wait_asynccnt)
#    define HAVE_WAIT_ASYNC_BUILTIN 1
#  endif
#endif

typedef __attribute__((ext_vector_type(4))) int v4i;
typedef __attribute__((address_space(1))) v4i gvec4i;   // global-AS 16B vector
typedef __attribute__((address_space(3))) v4i lvec4i;   // LDS-AS 16B vector

__device__ __forceinline__ void wait_asynccnt0() {
#if defined(USE_ASYNC_LDS)
#  if defined(HAVE_WAIT_ASYNC_BUILTIN)
    __builtin_amdgcn_s_wait_asynccnt(0);
#  else
    asm volatile("s_wait_asynccnt 0" ::: "memory");
#  endif
#endif
}

// 16B global -> LDS copy: async (no VGPR transit) if available
__device__ __forceinline__ void copy16_g2l(const unsigned short* g, unsigned short* l) {
#if defined(USE_ASYNC_LDS)
    __builtin_amdgcn_global_load_async_to_lds_b128(
        (gvec4i*)g, (lvec4i*)l, 0, 0);
#else
    *reinterpret_cast<uint4*>(l) = *reinterpret_cast<const uint4*>(g);
#endif
}

// ---------------- WMMA types ----------------
typedef __attribute__((ext_vector_type(16))) __bf16 v16bf;
typedef __attribute__((ext_vector_type(8)))  float  v8f;

union Frag16 { v16bf v; uint4 q[2]; };
union Acc8   { v8f v; float f[8]; };

// ---------------- bf16 helpers (RNE) ----------------
__device__ __forceinline__ unsigned bf16_rne_bits(float x) {
    unsigned u = __float_as_uint(x);
    return u + 0x7FFFu + ((u >> 16) & 1u);
}
__device__ __forceinline__ unsigned short f2bf(float x) {
    return (unsigned short)(bf16_rne_bits(x) >> 16);
}
__device__ __forceinline__ unsigned pk2(float a, float b) {
    return (bf16_rne_bits(a) >> 16) | (bf16_rne_bits(b) & 0xFFFF0000u);
}
__device__ __forceinline__ uint4 pk8(float4 f0, float4 f1) {
    uint4 r;
    r.x = pk2(f0.x, f0.y); r.y = pk2(f0.z, f0.w);
    r.z = pk2(f1.x, f1.y); r.w = pk2(f1.z, f1.w);
    return r;
}

// =====================================================================
// Kernel 1: router. One wave32 per token: logits (8), top-2, softmax.
// =====================================================================
__global__ __launch_bounds__(256) void router_kernel(
    const float* __restrict__ x, const float* __restrict__ rw,
    int* __restrict__ pair_e, float* __restrict__ pair_p)
{
    const int lane = threadIdx.x & 31;
    const int wid  = threadIdx.x >> 5;
    const int t = blockIdx.x * 8 + wid;
    const float* xp = x + (size_t)t * H_DIM;

    float acc[E_DIM];
#pragma unroll
    for (int e = 0; e < E_DIM; e++) acc[e] = 0.f;

    for (int h = lane; h < H_DIM; h += 32) {
        float xv = xp[h];
#pragma unroll
        for (int e = 0; e < E_DIM; e++) acc[e] += xv * rw[e * H_DIM + h];
    }
#pragma unroll
    for (int e = 0; e < E_DIM; e++) {
#pragma unroll
        for (int off = 16; off > 0; off >>= 1)
            acc[e] += __shfl_xor(acc[e], off, 32);
    }
    if (lane == 0) {
        int e0 = 0; float l0 = acc[0];
#pragma unroll
        for (int e = 1; e < E_DIM; e++) if (acc[e] > l0) { l0 = acc[e]; e0 = e; }
        int e1 = -1; float l1 = -INFINITY;
#pragma unroll
        for (int e = 0; e < E_DIM; e++) if (e != e0 && acc[e] > l1) { l1 = acc[e]; e1 = e; }
        // softmax over {l0, l1}, l0 is the max
        float p1 = expf(l1 - l0);
        float s = 1.0f + p1;
        pair_e[t * 2 + 0] = e0; pair_e[t * 2 + 1] = e1;
        pair_p[t * 2 + 0] = 1.0f / s; pair_p[t * 2 + 1] = p1 / s;
    }
}

// =====================================================================
// Kernel 2: deterministic bucketing (single block, no atomics).
// =====================================================================
__global__ __launch_bounds__(256) void bucket_kernel(
    const int* __restrict__ pair_e, const float* __restrict__ pair_p,
    int* __restrict__ counts, int* __restrict__ bases,
    int* __restrict__ tok_for_slot, float* __restrict__ prob_for_slot)
{
    __shared__ int hist[256][E_DIM];
    __shared__ int cur[256][E_DIM];
    __shared__ int tot[E_DIM];
    __shared__ int ebase[E_DIM];
    const int tid = threadIdx.x;
    const int p0 = tid * (NPAIR / 256);   // 32 pairs per thread

#pragma unroll
    for (int e = 0; e < E_DIM; e++) hist[tid][e] = 0;
    for (int i = 0; i < NPAIR / 256; i++) hist[tid][pair_e[p0 + i]]++;
    __syncthreads();

    if (tid < E_DIM) {
        int s = 0;
        for (int r = 0; r < 256; r++) s += hist[r][tid];
        tot[tid] = s;
        counts[tid] = s;
    }
    __syncthreads();
    if (tid == 0) {
        int b = 0;
        for (int e = 0; e < E_DIM; e++) { ebase[e] = b; bases[e] = b; b += tot[e]; }
    }
    __syncthreads();

#pragma unroll
    for (int e = 0; e < E_DIM; e++) {
        int s = ebase[e];
        for (int r = 0; r < tid; r++) s += hist[r][e];
        cur[tid][e] = s;
    }
    for (int i = 0; i < NPAIR / 256; i++) {
        int p = p0 + i;
        int e = pair_e[p];
        int slot = cur[tid][e]++;
        tok_for_slot[slot]  = p >> 1;      // token id
        prob_for_slot[slot] = pair_p[p];
    }
}

// =====================================================================
// Kernel 3: gather x rows into bf16 xg[slot, H]. One block per slot.
// =====================================================================
__global__ __launch_bounds__(256) void gather_kernel(
    const float* __restrict__ x, const int* __restrict__ tok_for_slot,
    unsigned short* __restrict__ xg)
{
    const int slot = blockIdx.x;
    const int t = tok_for_slot[slot];
    const float4* src = reinterpret_cast<const float4*>(x + (size_t)t * H_DIM);
    float4 f = src[threadIdx.x];                         // 256 thr * 4 = 1024
    uint2 o; o.x = pk2(f.x, f.y); o.y = pk2(f.z, f.w);
    reinterpret_cast<uint2*>(xg + (size_t)slot * H_DIM)[threadIdx.x] = o;
}

// =====================================================================
// Tiled bf16 WMMA GEMM: block 256 thr (8 waves), tile 128x128, K-step 64.
// Wave = 32(M) x 64(N) = 2x4 16x16 accumulators.
// LDS: [row][K] bf16, padded stride 72 elems (144B, 16B aligned).
// A tile staged via CDNA5 async global->LDS (bf16 passthrough);
// B tile staged via VGPRs with fp32->bf16 pack.
// =====================================================================
#define BM 128
#define BN 128
#define BK 64
#define LDK 72

// stage A tile (bf16 source, element stride srcld), 128 rows x 64 elems
__device__ __forceinline__ void stage_a(
    unsigned short* As, const unsigned short* src, size_t row0elem, int srcld,
    int kb, int tid)
{
#pragma unroll
    for (int i = 0; i < 4; i++) {
        int c = tid + 256 * i;
        int row = c >> 3, col = (c & 7) * 8;
        const unsigned short* g = src + row0elem + (size_t)row * srcld + kb + col;
        copy16_g2l(g, &As[row * LDK + col]);
    }
}

// stage B tile (fp32 source -> bf16), 128 rows x 64 elems; prefetch next k
__device__ __forceinline__ void stage_b(
    unsigned short* Bs, const float* bsrc, int srcld, int kb, int kb_next,
    int tid)
{
#pragma unroll
    for (int i = 0; i < 4; i++) {
        int c = tid + 256 * i;
        int row = c >> 3, col = (c & 7) * 8;
        const float* g = bsrc + (size_t)row * srcld + kb + col;
        const float4* s = reinterpret_cast<const float4*>(g);
        // one prefetch per 128B cacheline of the next K-stage
        if ((tid & 3) == 0)
            __builtin_prefetch(bsrc + (size_t)row * srcld + kb_next + col, 0, 1);
        *reinterpret_cast<uint4*>(&Bs[row * LDK + col]) = pk8(s[0], s[1]);
    }
}

// GEMM1: hmid[slot, f] = gelu( xg[slot,:] . w1[e,f,:] + b1[e,f] ), bf16 out
__global__ __launch_bounds__(256) void gemm1_kernel(
    const unsigned short* __restrict__ xg, const float* __restrict__ w1,
    const float* __restrict__ b1, unsigned short* __restrict__ hmid,
    const int* __restrict__ counts, const int* __restrict__ bases)
{
    __shared__ unsigned short As[BM * LDK];
    __shared__ unsigned short Bs[BN * LDK];

    const int e   = blockIdx.z;
    const int cnt = counts[e];
    const int mt  = blockIdx.x;
    if (mt * BM >= cnt) return;
    const int base = bases[e];
    const int nt   = blockIdx.y;

    const int tid  = threadIdx.x;
    const int lane = tid & 31, wid = tid >> 5;
    const int wm = (wid & 3) * 32;     // wave M offset within tile
    const int wn = (wid >> 2) * 64;    // wave N offset within tile

    const size_t arow0 = (size_t)(base + mt * BM) * H_DIM;
    const float* bsrc = w1 + (size_t)e * F_DIM * H_DIM + (size_t)(nt * BN) * H_DIM;

    Acc8 acc[8];
#pragma unroll
    for (int i = 0; i < 8; i++)
#pragma unroll
        for (int v = 0; v < 8; v++) acc[i].f[v] = 0.f;

    for (int kb = 0; kb < H_DIM; kb += BK) {
        int kb_next = (kb + BK < H_DIM) ? (kb + BK) : kb;
        stage_a(As, xg, arow0, H_DIM, kb, tid);
        stage_b(Bs, bsrc, H_DIM, kb, kb_next, tid);
        wait_asynccnt0();
        __syncthreads();

#pragma unroll
        for (int ks = 0; ks < BK; ks += 32) {
            Frag16 a[2], b[4];
            const int ka = ks + ((lane < 16) ? 0 : 8);
#pragma unroll
            for (int mi = 0; mi < 2; mi++) {
                const unsigned short* p = &As[(wm + mi * 16 + (lane & 15)) * LDK + ka];
                a[mi].q[0] = *reinterpret_cast<const uint4*>(p);
                a[mi].q[1] = *reinterpret_cast<const uint4*>(p + 16);
            }
            const int kbo = ks + ((lane < 16) ? 0 : 8);
#pragma unroll
            for (int ni = 0; ni < 4; ni++) {
                const unsigned short* p = &Bs[(wn + ni * 16 + (lane & 15)) * LDK + kbo];
                b[ni].q[0] = *reinterpret_cast<const uint4*>(p);
                b[ni].q[1] = *reinterpret_cast<const uint4*>(p + 16);
            }
#pragma unroll
            for (int mi = 0; mi < 2; mi++)
#pragma unroll
                for (int ni = 0; ni < 4; ni++)
                    acc[mi * 4 + ni].v = __builtin_amdgcn_wmma_f32_16x16x32_bf16(
                        false, a[mi].v, false, b[ni].v, (short)0, acc[mi * 4 + ni].v,
                        false, false);
        }
        __syncthreads();
    }

    // epilogue: +bias, exact-erf gelu, bf16 store (row-guarded)
    float bias[4];
#pragma unroll
    for (int ni = 0; ni < 4; ni++)
        bias[ni] = b1[e * F_DIM + nt * BN + wn + ni * 16 + (lane & 15)];

#pragma unroll
    for (int mi = 0; mi < 2; mi++) {
#pragma unroll
        for (int v = 0; v < 8; v++) {
            int m = mt * BM + wm + mi * 16 + ((lane >= 16) ? 8 : 0) + v;
            if (m >= cnt) continue;
            size_t rowp = (size_t)(base + m) * F_DIM;
#pragma unroll
            for (int ni = 0; ni < 4; ni++) {
                int n = nt * BN + wn + ni * 16 + (lane & 15);
                float xv = acc[mi * 4 + ni].f[v] + bias[ni];
                float g = 0.5f * xv * (1.0f + erff(xv * 0.70710678118654752f));
                hmid[rowp + n] = f2bf(g);
            }
        }
    }
}

// GEMM2: out[tok, h] += prob * ( hmid[slot,:] . w2[e,h,:] + b2[e,h] )
__global__ __launch_bounds__(256) void gemm2_kernel(
    const unsigned short* __restrict__ hmid, const float* __restrict__ w2,
    const float* __restrict__ b2, float* __restrict__ out,
    const int* __restrict__ counts, const int* __restrict__ bases,
    const int* __restrict__ tok_for_slot, const float* __restrict__ prob_for_slot)
{
    __shared__ unsigned short As[BM * LDK];
    __shared__ unsigned short Bs[BN * LDK];

    const int e   = blockIdx.z;
    const int cnt = counts[e];
    const int mt  = blockIdx.x;
    if (mt * BM >= cnt) return;
    const int base = bases[e];
    const int nt   = blockIdx.y;

    const int tid  = threadIdx.x;
    const int lane = tid & 31, wid = tid >> 5;
    const int wm = (wid & 3) * 32;
    const int wn = (wid >> 2) * 64;

    const size_t arow0 = (size_t)(base + mt * BM) * F_DIM;
    const float* bsrc = w2 + (size_t)e * H_DIM * F_DIM + (size_t)(nt * BN) * F_DIM;

    Acc8 acc[8];
#pragma unroll
    for (int i = 0; i < 8; i++)
#pragma unroll
        for (int v = 0; v < 8; v++) acc[i].f[v] = 0.f;

    for (int kb = 0; kb < F_DIM; kb += BK) {
        int kb_next = (kb + BK < F_DIM) ? (kb + BK) : kb;
        stage_a(As, hmid, arow0, F_DIM, kb, tid);
        stage_b(Bs, bsrc, F_DIM, kb, kb_next, tid);
        wait_asynccnt0();
        __syncthreads();

#pragma unroll
        for (int ks = 0; ks < BK; ks += 32) {
            Frag16 a[2], b[4];
            const int ka = ks + ((lane < 16) ? 0 : 8);
#pragma unroll
            for (int mi = 0; mi < 2; mi++) {
                const unsigned short* p = &As[(wm + mi * 16 + (lane & 15)) * LDK + ka];
                a[mi].q[0] = *reinterpret_cast<const uint4*>(p);
                a[mi].q[1] = *reinterpret_cast<const uint4*>(p + 16);
            }
            const int kbo = ks + ((lane < 16) ? 0 : 8);
#pragma unroll
            for (int ni = 0; ni < 4; ni++) {
                const unsigned short* p = &Bs[(wn + ni * 16 + (lane & 15)) * LDK + kbo];
                b[ni].q[0] = *reinterpret_cast<const uint4*>(p);
                b[ni].q[1] = *reinterpret_cast<const uint4*>(p + 16);
            }
#pragma unroll
            for (int mi = 0; mi < 2; mi++)
#pragma unroll
                for (int ni = 0; ni < 4; ni++)
                    acc[mi * 4 + ni].v = __builtin_amdgcn_wmma_f32_16x16x32_bf16(
                        false, a[mi].v, false, b[ni].v, (short)0, acc[mi * 4 + ni].v,
                        false, false);
        }
        __syncthreads();
    }

    float bias[4];
#pragma unroll
    for (int ni = 0; ni < 4; ni++)
        bias[ni] = b2[e * H_DIM + nt * BN + wn + ni * 16 + (lane & 15)];

#pragma unroll
    for (int mi = 0; mi < 2; mi++) {
#pragma unroll
        for (int v = 0; v < 8; v++) {
            int m = mt * BM + wm + mi * 16 + ((lane >= 16) ? 8 : 0) + v;
            if (m >= cnt) continue;
            int slot = base + m;
            int tok = tok_for_slot[slot];
            float pr = prob_for_slot[slot];
            float* orow = out + (size_t)tok * H_DIM;
#pragma unroll
            for (int ni = 0; ni < 4; ni++) {
                int n = nt * BN + wn + ni * 16 + (lane & 15);
                float val = (acc[mi * 4 + ni].f[v] + bias[ni]) * pr;
                unsafeAtomicAdd(&orow[n], val);   // exactly 2 adds/elem -> deterministic
            }
        }
    }
}

// ---------------- workspace layout (bytes) ----------------
static constexpr size_t OFF_PAIR_E = 0;                          // 8192 int
static constexpr size_t OFF_PAIR_P = 32768;                      // 8192 float
static constexpr size_t OFF_COUNTS = 65536;                      // 8 int
static constexpr size_t OFF_BASES  = 65536 + 64;                 // 8 int
static constexpr size_t OFF_TOK    = 65536 + 128;                // 8192 int
static constexpr size_t OFF_SLOTP  = OFF_TOK + 32768;            // 8192 float
static constexpr size_t OFF_XG     = 262144;                     // 8192*1024 bf16 = 16 MB
static constexpr size_t OFF_HMID   = OFF_XG + (size_t)NPAIR * H_DIM * 2; // 32 MB
// total ~50.6 MB (+ padded-tile overreads land inside hmid / slack)

extern "C" void kernel_launch(void* const* d_in, const int* in_sizes, int n_in,
                              void* d_out, int out_size, void* d_ws, size_t ws_size,
                              hipStream_t stream) {
    const float* x  = (const float*)d_in[0];   // [S,B,H] = [T,H]
    const float* rw = (const float*)d_in[1];   // [E,H]
    const float* w1 = (const float*)d_in[2];   // [E,F,H]
    const float* b1 = (const float*)d_in[3];   // [E,F]
    const float* w2 = (const float*)d_in[4];   // [E,H,F]
    const float* b2 = (const float*)d_in[5];   // [E,H]
    float* out = (float*)d_out;

    char* ws = (char*)d_ws;
    int*   pair_e = (int*)  (ws + OFF_PAIR_E);
    float* pair_p = (float*)(ws + OFF_PAIR_P);
    int*   counts = (int*)  (ws + OFF_COUNTS);
    int*   bases  = (int*)  (ws + OFF_BASES);
    int*   tok_fs = (int*)  (ws + OFF_TOK);
    float* prob_fs= (float*)(ws + OFF_SLOTP);
    unsigned short* xg   = (unsigned short*)(ws + OFF_XG);
    unsigned short* hmid = (unsigned short*)(ws + OFF_HMID);

    (void)hipMemsetAsync(d_out, 0, sizeof(float) * (size_t)T_DIM * H_DIM, stream);

    router_kernel<<<T_DIM / 8, 256, 0, stream>>>(x, rw, pair_e, pair_p);
    bucket_kernel<<<1, 256, 0, stream>>>(pair_e, pair_p, counts, bases, tok_fs, prob_fs);
    gather_kernel<<<NPAIR, 256, 0, stream>>>(x, tok_fs, xg);

    // per-expert count <= T=4096 -> at most 32 M-tiles of 128
    gemm1_kernel<<<dim3(32, F_DIM / BN, E_DIM), 256, 0, stream>>>(
        xg, w1, b1, hmid, counts, bases);
    gemm2_kernel<<<dim3(32, H_DIM / BN, E_DIM), 256, 0, stream>>>(
        hmid, w2, b2, out, counts, bases, tok_fs, prob_fs);
}
